// MultiHeadAttention_13297218748521
// MI455X (gfx1250) — compile-verified, matrix operand order checked by reading, not run on hardware
//
#include <hip/hip_runtime.h>
#include <hip/hip_bf16.h>

// ---------------------------------------------------------------------------
// Problem constants (match reference)
// ---------------------------------------------------------------------------
#define B_   2
#define S_   2048
#define E_   1024
#define H_   16
#define HD_  64
#define MD_  1024   // H_*HD_

typedef __bf16 bf16;
typedef __attribute__((ext_vector_type(16))) __bf16 v16bf;
typedef __attribute__((ext_vector_type(8)))  __bf16 bf16x8;
typedef __attribute__((ext_vector_type(8)))  float  v8f;
typedef __attribute__((ext_vector_type(8)))  float  f32x8;

// f32 -> bf16, round-to-nearest-even (used by the conversion, projection and attention kernels)
__device__ __forceinline__ bf16 f2bf(float f) {
    unsigned u = __builtin_bit_cast(unsigned, f);
    u += 0x7FFFu + ((u >> 16) & 1u);
    unsigned short h = (unsigned short)(u >> 16);
    return __builtin_bit_cast(bf16, h);
}

// ---------------------------------------------------------------------------
// WMMA fragment loader (CDNA5 16-bit 16x32 A layout; B assumed N-major mirror)
// Lane L<16 holds row m=L with K in {half..half+7} U {16+half..16+half+7},
// half = 0 for lanes 0-15, 8 for lanes 16-31.  Two 16B contiguous loads/lane.
// rowptr must already include the row (m = lane&15) and K-base offset.
// ---------------------------------------------------------------------------
__device__ __forceinline__ v16bf load16_bf(const bf16* rowptr, int lane) {
    const int half = (lane >> 4) << 3;                 // 0 or 8
    bf16x8 lo = *(const bf16x8*)(rowptr + half);       // k = half..half+7
    bf16x8 hi = *(const bf16x8*)(rowptr + 16 + half);  // k = 16+half..
    v16bf r;
#pragma unroll
    for (int i = 0; i < 8; ++i) { r[i] = lo[i]; r[8 + i] = hi[i]; }
    return r;
}

// cross-lane reductions inside each 16-lane half (wave32)
__device__ __forceinline__ float hmax16(float v) {
#pragma unroll
    for (int m = 1; m < 16; m <<= 1) v = fmaxf(v, __shfl_xor(v, m, 32));
    return v;
}
__device__ __forceinline__ float hsum16(float v) {
#pragma unroll
    for (int m = 1; m < 16; m <<= 1) v += __shfl_xor(v, m, 32);
    return v;
}

#define WMMA_BF16(A, Bv, C) \
    __builtin_amdgcn_wmma_f32_16x16x32_bf16(false, (A), false, (Bv), (short)0, (C), false, false)

// ---------------------------------------------------------------------------
// Async global -> LDS copy of one 64-key K/V tile (CDNA5 ASYNCcnt path).
// K tile: 64 rows x 64 bf16 contiguous (8 KB).  V tile: 64 d-rows x 128 B
// segments from Vt[d][t0..t0+63] (8 KB, LDS row stride 64 elems).
// 32 x global_load_async_to_lds_b128: each moves 32 lanes x 16 B = 512 B.
// ---------------------------------------------------------------------------
__device__ __forceinline__ void async_copy_tile(const bf16* Kg,      // tile start
                                                const bf16* Vg,      // head base [d][t]
                                                int t0,
                                                bf16* KldsB, bf16* VldsB,
                                                int lane) {
    // K: contiguous 8 KB
#pragma unroll
    for (int j = 0; j < 16; ++j) {
        unsigned q = (unsigned)(j * 512 + lane * 16);
        unsigned lds = (unsigned)(size_t)((const char*)KldsB + q);
        const char* g = (const char*)Kg + q;
        asm volatile("global_load_async_to_lds_b128 %0, %1, off"
                     :: "v"(lds), "v"(g) : "memory");
    }
    // V: 64 rows, 128 B per row; 4 rows per instruction
    const int vr = lane >> 3;            // 0..3 row-in-group
    const int vb = (lane & 7) * 16;      // byte within 128B row segment
#pragma unroll
    for (int j = 0; j < 16; ++j) {
        int d = j * 4 + vr;
        unsigned lds = (unsigned)(size_t)((const char*)VldsB + d * 128 + vb);
        const char* g = (const char*)Vg + (size_t)d * (S_ * 2) + (size_t)t0 * 2 + vb;
        asm volatile("global_load_async_to_lds_b128 %0, %1, off"
                     :: "v"(lds), "v"(g) : "memory");
    }
}

// ---------------------------------------------------------------------------
// Kernel: bulk f32 -> bf16, 8 elements per thread (memory-bound one-time pass)
// ---------------------------------------------------------------------------
__global__ __launch_bounds__(256)
void cvt_bf16_kernel(const float* __restrict__ in, bf16* __restrict__ out, int n8) {
    int i = blockIdx.x * blockDim.x + threadIdx.x;
    if (i >= n8) return;
    f32x8 v = *(const f32x8*)(in + (size_t)i * 8);
    bf16x8 o;
#pragma unroll
    for (int j = 0; j < 8; ++j) o[j] = f2bf(v[j]);
    *(bf16x8*)(out + (size_t)i * 8) = o;
}

// ---------------------------------------------------------------------------
// Kernel: transpose + convert weights  in[batch][M][N] f32 -> out[batch][N][M] bf16
// ---------------------------------------------------------------------------
__global__ void transpose_cvt_kernel(const float* __restrict__ in,
                                     bf16* __restrict__ out, int M, int N) {
    int idx = blockIdx.x * blockDim.x + threadIdx.x;
    if (idx >= M * N) return;
    int m = idx / N, n = idx % N;
    const float* ib = in + (size_t)blockIdx.z * M * N;
    bf16*        ob = out + (size_t)blockIdx.z * M * N;
    ob[(size_t)n * M + m] = f2bf(ib[idx]);
}

// ---------------------------------------------------------------------------
// Kernel: per-head projection  Out[b,h,s,d] = X[b,s,:] . W[h,:,d] + bias[h,d]
// Xbf is bf16 [B][S][E]; WT is pre-transposed bf16 [H][HD][E].
// 4 waves/block, each wave one 16-wide N-tile of HD.
// transposeOut=1 stores Out as [b,h,d,s] (used for V).
// ---------------------------------------------------------------------------
__global__ __launch_bounds__(128)
void qkv_proj_kernel(const bf16* __restrict__ Xbf, const bf16* __restrict__ WT,
                     const float* __restrict__ bias, bf16* __restrict__ Out,
                     int transposeOut) {
    const int lane = threadIdx.x & 31;
    const int wave = threadIdx.x >> 5;        // n-tile: 0..3
    const int mt = blockIdx.x, h = blockIdx.y, b = blockIdx.z;
    const int m  = lane & 15;
    const int s0 = mt * 16;

    const bf16* Xr = Xbf + ((size_t)b * S_ + s0 + m) * E_;
    const bf16* Wr = WT  + ((size_t)h * HD_ + wave * 16 + m) * E_;

    v8f c = {};
#pragma unroll 4
    for (int e0 = 0; e0 < E_; e0 += 32) {
        v16bf a  = load16_bf(Xr + e0, lane);
        v16bf bw = load16_bf(Wr + e0, lane);
        c = WMMA_BF16(a, bw, c);
    }

    const int   col  = wave * 16 + (lane & 15);
    const float bv   = bias[h * HD_ + col];
    const int   row0 = (lane >> 4) << 3;       // 0 or 8

    if (!transposeOut) {
        const size_t base = (size_t)(b * H_ + h) * S_;
#pragma unroll
        for (int r = 0; r < 8; ++r) {
            int row = s0 + r + row0;
            Out[(base + row) * HD_ + col] = f2bf(c[r] + bv);
        }
    } else {
        const size_t base = ((size_t)(b * H_ + h) * HD_ + col) * S_;
#pragma unroll
        for (int r = 0; r < 8; ++r)
            Out[base + s0 + r + row0] = f2bf(c[r] + bv);
    }
}

// ---------------------------------------------------------------------------
// Kernel: flash-attention per (b,h, 16-row Q tile).  One wave per block.
// 64-key steps, double-buffered async global->LDS staging of K/V tiles
// (ASYNCcnt), 16 WMMAs per iteration.  Q,K bf16 [b,h,s,d]; Vt bf16 [b,h,d,t];
// ctx bf16 [b,s,h*HD]
// ---------------------------------------------------------------------------
__global__ __launch_bounds__(32)
void attention_kernel(const bf16* __restrict__ Q, const bf16* __restrict__ K,
                      const bf16* __restrict__ Vt, bf16* __restrict__ ctx) {
    const int lane = threadIdx.x;
    const int mt = blockIdx.x, h = blockIdx.y, b = blockIdx.z;
    const size_t bh = (size_t)(b * H_ + h);
    const bf16* Qb = Q  + bh * S_ * HD_;
    const bf16* Kb = K  + bh * S_ * HD_;
    const bf16* Vb = Vt + bh * HD_ * S_;     // [d][t]
    const int m  = lane & 15;
    const int s0 = mt * 16;
    const int row0 = (lane >> 4) << 3;       // 0 or 8

    const v16bf aQ0 = load16_bf(Qb + (size_t)(s0 + m) * HD_ + 0,  lane);
    const v16bf aQ1 = load16_bf(Qb + (size_t)(s0 + m) * HD_ + 32, lane);

    v8f oacc[4] = {{}, {}, {}, {}};
    float mrun[8], lrun[8];
#pragma unroll
    for (int r = 0; r < 8; ++r) { mrun[r] = -1e30f; lrun[r] = 0.f; }

    __shared__ __align__(16) bf16 Klds[2][64 * 64];   // 2 x 8 KB
    __shared__ __align__(16) bf16 Vlds[2][64 * 64];   // 2 x 8 KB
    __shared__ __align__(16) bf16 Plds[16 * 64];      // 2 KB

    const int NT = S_ / 64;                  // 32 tiles
    async_copy_tile(Kb, Vb, 0, Klds[0], Vlds[0], lane);

    for (int i = 0; i < NT; ++i) {
        const bf16* Kc = Klds[i & 1];
        const bf16* Vc = Vlds[i & 1];

        if (i + 1 < NT) {
            // fence prior LDS reads of the buffer we are about to refill,
            // then issue next tile's async copies and wait for current tile
            asm volatile("s_wait_dscnt 0x0" ::: "memory");
            async_copy_tile(Kb + (size_t)(i + 1) * 64 * HD_, Vb, (i + 1) * 64,
                            Klds[(i + 1) & 1], Vlds[(i + 1) & 1], lane);
            asm volatile("s_wait_asynccnt 0x20" ::: "memory");  // <=32 left
        } else {
            asm volatile("s_wait_asynccnt 0x0" ::: "memory");
        }

        // scores: four 16x16 column tiles, K-dim 64 = 2 x WMMA(K=32) each
        v8f sc[4];
#pragma unroll
        for (int j = 0; j < 4; ++j) {
            v16bf blo = load16_bf(Kc + (size_t)(j * 16 + m) * 64 + 0,  lane);
            v16bf bhi = load16_bf(Kc + (size_t)(j * 16 + m) * 64 + 32, lane);
            v8f s = {};
            s = WMMA_BF16(aQ0, blo, s);
            s = WMMA_BF16(aQ1, bhi, s);
            sc[j] = s;
        }

        // online softmax update; row of element r is (r | row0) per lane half
#pragma unroll
        for (int r = 0; r < 8; ++r) {
            float x0 = sc[0][r] * 0.125f;        // 1/sqrt(64)
            float x1 = sc[1][r] * 0.125f;
            float x2 = sc[2][r] * 0.125f;
            float x3 = sc[3][r] * 0.125f;
            float tmax  = hmax16(fmaxf(fmaxf(x0, x1), fmaxf(x2, x3)));
            float mnew  = fmaxf(mrun[r], tmax);
            float alpha = __expf(mrun[r] - mnew);
            float p0 = __expf(x0 - mnew);
            float p1 = __expf(x1 - mnew);
            float p2 = __expf(x2 - mnew);
            float p3 = __expf(x3 - mnew);
            lrun[r] = lrun[r] * alpha + hsum16((p0 + p1) + (p2 + p3));
            mrun[r] = mnew;
            oacc[0][r] *= alpha; oacc[1][r] *= alpha;
            oacc[2][r] *= alpha; oacc[3][r] *= alpha;
            bf16* Pr = &Plds[(r + row0) * 64 + (lane & 15)];
            Pr[0]  = f2bf(p0);
            Pr[16] = f2bf(p1);
            Pr[32] = f2bf(p2);
            Pr[48] = f2bf(p3);
        }
        __syncthreads();                              // single-wave: S_NOP-cheap
        v16bf aPlo = load16_bf(&Plds[(lane & 15) * 64 +  0], lane);
        v16bf aPhi = load16_bf(&Plds[(lane & 15) * 64 + 32], lane);
        __syncthreads();

        // P(16x64) x V(64x64): 4 N-tiles x 2 K-halves from the LDS V tile
#pragma unroll
        for (int j = 0; j < 4; ++j) {
            v16bf bvlo = load16_bf(Vc + (size_t)(j * 16 + m) * 64 + 0,  lane);
            v16bf bvhi = load16_bf(Vc + (size_t)(j * 16 + m) * 64 + 32, lane);
            oacc[j] = WMMA_BF16(aPlo, bvlo, oacc[j]);
            oacc[j] = WMMA_BF16(aPhi, bvhi, oacc[j]);
        }
    }

    // normalize + store concat-head context (bf16)
#pragma unroll
    for (int r = 0; r < 8; ++r) {
        float inv = 1.0f / lrun[r];
        int row = s0 + r + row0;
        size_t base = ((size_t)b * S_ + row) * MD_ + (size_t)h * HD_;
        ctx[base +  0 + (lane & 15)] = f2bf(oacc[0][r] * inv);
        ctx[base + 16 + (lane & 15)] = f2bf(oacc[1][r] * inv);
        ctx[base + 32 + (lane & 15)] = f2bf(oacc[2][r] * inv);
        ctx[base + 48 + (lane & 15)] = f2bf(oacc[3][r] * inv);
    }
}

// ---------------------------------------------------------------------------
// Kernel: out = ctx @ Wo + bo.   ctx bf16 [B*S][MD], WoT bf16 [E][MD]
// 4 waves/block, each wave a 16-wide N-tile (block covers 64 of E)
// ---------------------------------------------------------------------------
__global__ __launch_bounds__(128)
void out_proj_kernel(const bf16* __restrict__ ctx, const bf16* __restrict__ WoT,
                     const float* __restrict__ bo, float* __restrict__ out) {
    const int lane = threadIdx.x & 31;
    const int wave = threadIdx.x >> 5;
    const int mt = blockIdx.x;                 // B*S / 16
    const int nb = blockIdx.y;                 // E / 64
    const int m  = lane & 15;
    const int n  = nb * 64 + wave * 16 + (lane & 15);

    const bf16* Ar = ctx + ((size_t)mt * 16 + m) * MD_;
    const bf16* Br = WoT + (size_t)n * MD_;

    v8f c = {};
#pragma unroll 4
    for (int k0 = 0; k0 < MD_; k0 += 32) {
        v16bf a  = load16_bf(Ar + k0, lane);
        v16bf bw = load16_bf(Br + k0, lane);
        c = WMMA_BF16(a, bw, c);
    }

    const float bb = bo[n];
    const int row0 = (lane >> 4) << 3;
#pragma unroll
    for (int r = 0; r < 8; ++r) {
        int row = mt * 16 + r + row0;
        out[(size_t)row * E_ + n] = c[r] + bb;
    }
}

// ---------------------------------------------------------------------------
// Host launcher
// ---------------------------------------------------------------------------
extern "C" void kernel_launch(void* const* d_in, const int* in_sizes, int n_in,
                              void* d_out, int out_size, void* d_ws, size_t ws_size,
                              hipStream_t stream) {
    const float* query = (const float*)d_in[0];
    const float* key   = (const float*)d_in[1];
    const float* value = (const float*)d_in[2];
    const float* Wq    = (const float*)d_in[3];
    const float* bq    = (const float*)d_in[4];
    const float* Wk    = (const float*)d_in[5];
    const float* bk    = (const float*)d_in[6];
    const float* Wv    = (const float*)d_in[7];
    const float* bv    = (const float*)d_in[8];
    const float* Wo    = (const float*)d_in[9];
    const float* bo    = (const float*)d_in[10];
    float* out = (float*)d_out;

    char* ws = (char*)d_ws;
    size_t off = 0;
    auto take = [&](size_t bytes) -> void* {
        void* p = ws + off;
        off += (bytes + 255) & ~(size_t)255;
        return p;
    };

    bf16* WqT = (bf16*)take((size_t)H_ * E_ * HD_ * sizeof(bf16));
    bf16* WkT = (bf16*)take((size_t)H_ * E_ * HD_ * sizeof(bf16));
    bf16* WvT = (bf16*)take((size_t)H_ * E_ * HD_ * sizeof(bf16));
    bf16* WoT = (bf16*)take((size_t)MD_ * E_ * sizeof(bf16));
    bf16* Xq  = (bf16*)take((size_t)B_ * S_ * E_ * sizeof(bf16));
    bf16* Xk  = (bf16*)take((size_t)B_ * S_ * E_ * sizeof(bf16));
    bf16* Xv  = (bf16*)take((size_t)B_ * S_ * E_ * sizeof(bf16));
    bf16* Qb  = (bf16*)take((size_t)B_ * H_ * S_ * HD_ * sizeof(bf16));
    bf16* Kb  = (bf16*)take((size_t)B_ * H_ * S_ * HD_ * sizeof(bf16));
    bf16* Vt  = (bf16*)take((size_t)B_ * H_ * S_ * HD_ * sizeof(bf16));
    bf16* ctx = (bf16*)take((size_t)B_ * S_ * MD_ * sizeof(bf16));
    (void)ws_size; (void)in_sizes; (void)n_in; (void)out_size;

    // 1) bulk activation conversion f32 -> bf16 (memory-bound, one pass each)
    {
        const int n8 = (B_ * S_ * E_) / 8;
        dim3 g((n8 + 255) / 256);
        cvt_bf16_kernel<<<g, 256, 0, stream>>>(query, Xq, n8);
        cvt_bf16_kernel<<<g, 256, 0, stream>>>(key,   Xk, n8);
        cvt_bf16_kernel<<<g, 256, 0, stream>>>(value, Xv, n8);
    }

    // 2) weight transpose+convert to bf16
    {
        dim3 g((E_ * HD_ + 255) / 256, 1, H_);
        transpose_cvt_kernel<<<g, 256, 0, stream>>>(Wq, WqT, E_, HD_);
        transpose_cvt_kernel<<<g, 256, 0, stream>>>(Wk, WkT, E_, HD_);
        transpose_cvt_kernel<<<g, 256, 0, stream>>>(Wv, WvT, E_, HD_);
        dim3 g2((MD_ * E_ + 255) / 256, 1, 1);
        transpose_cvt_kernel<<<g2, 256, 0, stream>>>(Wo, WoT, MD_, E_);
    }

    // 3) Q/K/V projections (V stored transposed for contiguous PV fragments)
    dim3 gp(S_ / 16, H_, B_);
    qkv_proj_kernel<<<gp, 128, 0, stream>>>(Xq, WqT, bq, Qb, 0);
    qkv_proj_kernel<<<gp, 128, 0, stream>>>(Xk, WkT, bk, Kb, 0);
    qkv_proj_kernel<<<gp, 128, 0, stream>>>(Xv, WvT, bv, Vt, 1);

    // 4) flash attention (64-key steps, async double-buffered K/V staging)
    attention_kernel<<<dim3(S_ / 16, H_, B_), 32, 0, stream>>>(Qb, Kb, Vt, ctx);

    // 5) output projection
    out_proj_kernel<<<dim3(B_ * S_ / 16, E_ / 64), 128, 0, stream>>>(ctx, WoT, bo, out);
}
